// RLGCNN_55920474194389
// MI455X (gfx1250) — compile-verified, matrix operand order checked by reading, not run on hardware
//
#include <hip/hip_runtime.h>

// ============================================================================
// Recurrent 7-layer 1x3-conv GNN on MI455X (gfx1250), WMMA f16 formulation.
//
// Per node n all 64 batches share neighbor indices.  Each wave owns 16
// batches and runs the 7-layer chain as banded matmuls:
//     Out(feat x batch) = Wfull(feat x K) @ In(K x batch)
// WMMA roles:  A = weights (documented ISA A-layout, staged once in LDS),
//              B = activations (K x 16 batches, per-wave LDS ping-pong),
//              D = feat x batch -> each lane holds 8 consecutive features of
//                  one batch, so the ReLU'd tile stores as ONE ds_store_b128.
// Bias is fused as an extra K row against a self-propagating ones feature
// (act[KDIM]=1, W[NFEA][KDIM]=1).  ReLU is done as packed-f16 max (identical
// to relu-then-convert since RNE preserves sign).  Zero padding propagates.
// Pass 4 evaluates only the driver nodes and writes d_out directly.
// Register pressure is capped (launch_bounds + per-layer sched barriers) to
// stay inside the 256-VGPR direct window: no s_set_vgpr_msb churn.
// ============================================================================

#define BATCH   64
#define DNB     16
#define NLAYER  7
#define BLOCK   128   // 4 waves of 32

// conv chain:        L0  L1  L2  L3  L4  L5  L6
static constexpr int CIa[NLAYER]  = { 1,  5,  5,  5,  5,  5,  5};
static constexpr int COa[NLAYER]  = { 5,  5,  5,  5,  5,  5,  1};
static constexpr int KDIM[NLAYER] = {16, 70, 60, 50, 40, 30, 20};  // CI*Win
static constexpr int NFEA[NLAYER] = {70, 60, 50, 40, 30, 20,  2};  // CO*Wout
static constexpr int KP[NLAYER]   = {32, 96, 64, 64, 64, 32, 32};  // K pad 32
static constexpr int NP[NLAYER]   = {96, 64, 64, 64, 32, 32, 16};  // == KP[l+1]
static constexpr int WOFF[NLAYER] = {0, 3072, 9216, 13312, 17408, 19456, 20480};
static constexpr int WTOT         = 20992;  // f16 elems (41 KB)

typedef __attribute__((ext_vector_type(16))) _Float16 v16h;
typedef __attribute__((ext_vector_type(2)))  _Float16 v2h;
typedef __attribute__((ext_vector_type(8)))  float    v8f;
typedef __attribute__((ext_vector_type(4)))  unsigned int v4u;

struct NetParams { const float* w[NLAYER]; const float* b[NLAYER]; };

__device__ __forceinline__ unsigned short f2h(float v) {
  _Float16 t = (_Float16)v;
  return __builtin_bit_cast(unsigned short, t);
}

// cvt two f32 to packed f16 dword, then ReLU as one v_pk_max_num_f16
__device__ __forceinline__ unsigned int pk_relu(float o0, float o1) {
  v2h p;
  p[0] = (_Float16)o0;
  p[1] = (_Float16)o1;
  v2h z = {};
  p = __builtin_elementwise_max(p, z);
  return __builtin_bit_cast(unsigned int, p);
}

// One 16-bit WMMA operand (this lane's 16 halves) from LDS: two 16B chunks.
// ISA 16-bit operand layout: lane-half h covers K {h*8..h*8+7} and
// {16+h*8..16+h*8+7} of each 32-wide K chunk (B assumed to mirror A).
__device__ __forceinline__ v16h ld_op(const unsigned short* base, int e0) {
  struct { v4u a, b; } t;
  t.a = *reinterpret_cast<const v4u*>(base + e0);
  t.b = *reinterpret_cast<const v4u*>(base + e0 + 16);
  return __builtin_bit_cast(v16h, t);
}

__device__ __forceinline__ v8f wmma16(v16h a, v16h b, v8f c) {
  return __builtin_amdgcn_wmma_f32_16x16x32_f16(false, a, false, b,
                                                (short)0, c, false, false);
}

// One layer for this wave's 16 batches.  h = lane>>4, m = lane&15.
// m is both the B/D batch column and the A feature row within a tile.
template <int L>
__device__ __forceinline__ void do_layer(const unsigned short* __restrict__ actIn,
                                         unsigned short* __restrict__ actOut,
                                         float* __restrict__ res2,
                                         const unsigned short* __restrict__ sw,
                                         int h, int m) {
  constexpr int KPL = KP[L];
  constexpr int KCH = KPL / 32;
  constexpr int NT  = NP[L] / 16;   // feature (M) tiles

  // B operand: this lane's batch column, all K chunks (reused across M tiles)
  v16h b[KCH];
#pragma unroll
  for (int kc = 0; kc < KCH; ++kc)
    b[kc] = ld_op(actIn + m * KPL, kc * 32 + h * 8);

#pragma unroll
  for (int t = 0; t < NT; ++t) {
    v8f acc = {};
#pragma unroll
    for (int kc = 0; kc < KCH; ++kc) {
      // A operand: weight rows (feature t*16+m), documented A layout
      v16h a = ld_op(sw + WOFF[L] + (t * 16 + m) * KPL, kc * 32 + h * 8);
      acc = wmma16(a, b[kc], acc);   // EXEC all-ones: no divergence here
    }
    if constexpr (L < NLAYER - 1) {
      // D: lane holds feats t*16+8h .. +7 of batch m -> one b128 store
      v4u vv = {pk_relu(acc[0], acc[1]), pk_relu(acc[2], acc[3]),
                pk_relu(acc[4], acc[5]), pk_relu(acc[6], acc[7])};
      *reinterpret_cast<v4u*>(actOut + m * KP[L + 1] + t * 16 + h * 8) = vv;
    } else {
      // final layer: lanes with h==0 hold feats 0,1 of batch m in-register
      res2[0] = acc[0] > 0.0f ? acc[0] : 0.0f;
      res2[1] = acc[1] > 0.0f ? acc[1] : 0.0f;
    }
  }
}

__global__ void __launch_bounds__(BLOCK, 4)
gnn_pass_kernel(const float* __restrict__ zin, float* __restrict__ zout,
                const int* __restrict__ nbr, NetParams P, int Nn,
                const int* __restrict__ nodeList, int nWork, int outPitch) {
  __shared__ unsigned short s_w[WTOT];             // banded weights + bias row
  __shared__ unsigned short s_act[4][2][96 * 16];  // per-wave ping-pong acts

  const int tid  = threadIdx.x;
  const int wave = tid >> 5;
  const int lane = tid & 31;
  const int h    = lane >> 4;   // K-half / feature-half selector
  const int m    = lane & 15;   // batch column == A feature row in tile

  // ---- expand banded Wfull (feat_out x K) into LDS, zero padded.
  //      Row k==KDIM[l] carries the bias (hit by the ones feature);
  //      feature row NFEA[l] regenerates the ones feature for the next layer.
  for (int l = 0; l < NLAYER; ++l) {
    const int kp = KP[l], tot = kp * NP[l];
    for (int i = tid; i < tot; i += BLOCK) {
      int col = i / kp, k = i % kp;   // col = output feature, k = input feature
      float v = 0.0f;
      if (col < NFEA[l]) {
        if (k < KDIM[l]) {
          int wo = col / COa[l], co = col % COa[l];
          int wi = k / CIa[l],  ci = k % CIa[l];
          int r = wi - wo;
          if (r >= 0 && r < 3) v = P.w[l][(co * CIa[l] + ci) * 3 + r];
        } else if (k == KDIM[l]) {
          v = P.b[l][col % COa[l]];          // fused bias row
        }
      } else if (col == NFEA[l] && k == KDIM[l]) {
        v = 1.0f;                            // ones-feature self-propagation
      }
      s_w[WOFF[l] + i] = f2h(v);
    }
  }
  __syncthreads();

  unsigned short* a0 = s_act[wave][0];
  unsigned short* a1 = s_act[wave][1];
  const int batchBase = wave * 16;

  for (int it = blockIdx.x; it < nWork; it += gridDim.x) {
    const int node = nodeList ? nodeList[it] : it;
    __builtin_prefetch(nbr + (size_t)(node + gridDim.x) * DNB, 0, 1);

    // ---- gather: a0[batch m][feat d] = z[16w+m][nbr[node][d]]
    //      feats 16..31 pad: feat 16 = 1.0 (ones feature), rest 0
    {
      const int* ni = nbr + (size_t)node * DNB;
      const int d0 = h * 8;
      float vals[8];
#pragma unroll
      for (int j = 0; j < 8; ++j) {
        int ii = ni[d0 + j];
        vals[j] = zin[(size_t)(batchBase + m) * Nn + ii];
      }
      unsigned int pk[4];
#pragma unroll
      for (int r = 0; r < 4; ++r)
        pk[r] = (unsigned)f2h(vals[2 * r]) | ((unsigned)f2h(vals[2 * r + 1]) << 16);
      v4u vv = {pk[0], pk[1], pk[2], pk[3]};
      *reinterpret_cast<v4u*>(a0 + m * KP[0] + d0) = vv;
      v4u pad = {h == 0 ? 0x3C00u : 0u, 0u, 0u, 0u};   // f16 1.0 at feat 16
      *reinterpret_cast<v4u*>(a0 + m * KP[0] + 16 + d0) = pad;
    }

    float r2[2];
    __builtin_amdgcn_sched_barrier(0);   // keep layers from interleaving:
    do_layer<0>(a0, a1, nullptr, s_w, h, m);  __builtin_amdgcn_sched_barrier(0);
    do_layer<1>(a1, a0, nullptr, s_w, h, m);  __builtin_amdgcn_sched_barrier(0);
    do_layer<2>(a0, a1, nullptr, s_w, h, m);  __builtin_amdgcn_sched_barrier(0);
    do_layer<3>(a1, a0, nullptr, s_w, h, m);  __builtin_amdgcn_sched_barrier(0);
    do_layer<4>(a0, a1, nullptr, s_w, h, m);  __builtin_amdgcn_sched_barrier(0);
    do_layer<5>(a1, a0, nullptr, s_w, h, m);  __builtin_amdgcn_sched_barrier(0);
    do_layer<6>(a0, nullptr, r2, s_w, h, m);
    __builtin_amdgcn_sched_barrier(0);

    if (h == 0) {   // lanes 0-15 hold feats 0,1 of their batch in-register
      float z = 0.5f * (r2[0] + r2[1]);
      zout[(size_t)(batchBase + m) * outPitch + it] = z;
    }
  }
}

extern "C" void kernel_launch(void* const* d_in, const int* in_sizes, int n_in,
                              void* d_out, int out_size, void* d_ws, size_t ws_size,
                              hipStream_t stream) {
  const float* x    = (const float*)d_in[0];
  const int*   nbr  = (const int*)d_in[1];
  const int*   didx = (const int*)d_in[2];
  const int Nn   = in_sizes[0] / BATCH;
  const int ndrv = in_sizes[2];

  NetParams P;
  for (int l = 0; l < NLAYER; ++l) {
    P.w[l] = (const float*)d_in[3 + 2 * l];
    P.b[l] = (const float*)d_in[4 + 2 * l];
  }

  // ping-pong z state in workspace (needs 2 * 64 * N * 4 bytes = ~25.6 MB)
  float* z1 = (float*)d_ws;
  float* z2 = z1 + (size_t)BATCH * Nn;

  dim3 block(BLOCK);
  // passes 1-3: all nodes
  gnn_pass_kernel<<<dim3(2048), block, 0, stream>>>(x,  z1, nbr, P, Nn,
                                                    nullptr, Nn, Nn);
  gnn_pass_kernel<<<dim3(2048), block, 0, stream>>>(z1, z2, nbr, P, Nn,
                                                    nullptr, Nn, Nn);
  gnn_pass_kernel<<<dim3(2048), block, 0, stream>>>(z2, z1, nbr, P, Nn,
                                                    nullptr, Nn, Nn);
  // pass 4: only the driver nodes, written straight into d_out (64 x ndrv)
  int g4 = ndrv < 2048 ? ndrv : 2048;
  gnn_pass_kernel<<<dim3(g4), block, 0, stream>>>(z1, (float*)d_out, nbr, P, Nn,
                                                  didx, ndrv, ndrv);
}
